// AttnEncoderXL_69836168233164
// MI455X (gfx1250) — matrix operand order repaired and checked by reading, not compiled, not run on hardware
//
#include <hip/hip_runtime.h>
#include <hip/hip_bf16.h>
#include <math.h>

// ---------------------------------------------------------------------------
// AttnEncoderXL for MI455X (gfx1250, wave32, WMMA).
//
// Sizes: B=8, T=256, D=256, H=8, DH=32, L=4, DFF=1024, N=B*T=2048.
//
//  * All GEMMs use v_wmma_f32_16x16x32_f16 (f16 in, f32 accumulate).
//  * Heads 4..7 relative bias is k-constant (temb broadcast) -> softmax
//    shift-invariant -> timestep-MLP path is dead code; skipped.
//  * rel tensor (536 MB) never materialized: RBF bias for heads 0..3 is
//    computed on the fly from bond_matrix inside the attention kernel.
//  * GEMM is software-pipelined with double-buffered LDS: next slab's global
//    loads issue before the WMMAs, LDS stores after them, 1 barrier/slab.
//  * LDS tiles stored in fragment order (B transposed) so every WMMA operand
//    is two aligned ds_load_b128s.
// ---------------------------------------------------------------------------

typedef __attribute__((ext_vector_type(16))) _Float16 v16h;
typedef __attribute__((ext_vector_type(8)))  _Float16 v8h;
typedef __attribute__((ext_vector_type(4)))  _Float16 v4h;
typedef __attribute__((ext_vector_type(8)))  float    v8f;

#define BB 8
#define TT 256
#define DD 256
#define HH 8
#define DHH 32
#define NN 2048
#define DFFF 1024

__device__ __forceinline__ v16h cat16(v8h lo, v8h hi) {
  return __builtin_shufflevector(lo, hi, 0, 1, 2, 3, 4, 5, 6, 7,
                                 8, 9, 10, 11, 12, 13, 14, 15);
}

__device__ __forceinline__ v4h cvt4(float4 f) {
  v4h h;
  h[0] = (_Float16)f.x; h[1] = (_Float16)f.y;
  h[2] = (_Float16)f.z; h[3] = (_Float16)f.w;
  return h;
}

// ---------------------------------------------------------------------------
// Embedding gather: x[n, :] = emb_table[src[n], :]
// ---------------------------------------------------------------------------
__global__ void embed_kernel(const int* __restrict__ src,
                             const float* __restrict__ emb,
                             float* __restrict__ x) {
  const int n = blockIdx.x;
  const int t = threadIdx.x;
  x[n * DD + t] = emb[src[n] * DD + t];
}

// ---------------------------------------------------------------------------
// LayerNorm: one block (256 threads) per row of [N, 256].
// ---------------------------------------------------------------------------
__global__ void ln_kernel(const float* __restrict__ x,
                          const float* __restrict__ g,
                          const float* __restrict__ bta,
                          float* __restrict__ out) {
  __shared__ float red[256];
  __shared__ float s_mean, s_rstd;
  const int row = blockIdx.x;
  const int t = threadIdx.x;
  float val = x[row * DD + t];
  red[t] = val;
  __syncthreads();
  for (int s = 128; s > 0; s >>= 1) {
    if (t < s) red[t] += red[t + s];
    __syncthreads();
  }
  if (t == 0) s_mean = red[0] * (1.0f / DD);
  __syncthreads();
  float d = val - s_mean;
  red[t] = d * d;
  __syncthreads();
  for (int s = 128; s > 0; s >>= 1) {
    if (t < s) red[t] += red[t + s];
    __syncthreads();
  }
  if (t == 0) s_rstd = rsqrtf(red[0] * (1.0f / DD) + 1e-6f);
  __syncthreads();
  out[row * DD + t] = d * s_rstd * g[t] + bta[t];
}

// ---------------------------------------------------------------------------
// Tiled WMMA GEMM: C[N,M] = epilogue(A[N,K] @ W[K,M] + bias).
// Block tile 128x64, 256 threads = 8 waves, wave w -> rows [16w,16w+16).
// modes: 0 = bias only, 1 = bias + residual, 2 = relu(bias).
// Software-pipelined: double-buffered LDS, global loads for slab t+1 issue
// before slab t's WMMAs, LDS stores after, one barrier per slab.
// ---------------------------------------------------------------------------
__global__ void __launch_bounds__(256)
gemm_kernel(const float* __restrict__ A,
            const float* __restrict__ W,
            const float* __restrict__ bias,
            const float* __restrict__ res,
            float* __restrict__ C,
            int N, int K, int M, int mode) {
  __shared__ _Float16 As[2][128 * 40];  // [row][k]  128 x 32, pad 40
  __shared__ _Float16 Bt[2][64 * 40];   // [col][k]   64 x 32, pad 40 (transposed)
  const int n0 = blockIdx.y * 128;
  const int m0 = blockIdx.x * 64;
  const int tid = threadIdx.x;
  const int lane = tid & 31;
  const int wv = tid >> 5;

  v8f acc[4];
  for (int ct = 0; ct < 4; ct++)
    for (int i = 0; i < 8; i++) acc[ct][i] = 0.0f;

  // Fragment geometry: element s of lane l holds k = 16*(l/16) + s, same order for A and B.
  const int arow = wv * 16 + (lane & 15);
  const int akoff = (lane < 16) ? 0 : 16;
  const int bcol = lane & 15;
  const int bkoff = (lane < 16) ? 0 : 16;

  float4 a4[4], w4[2];

  // Issue all global loads for slab k0 (no waits interleaved).
  auto gload = [&](int k0) {
#pragma unroll
    for (int i = 0; i < 4; i++) {
      int idx = i * 256 + tid;           // 1024 float4s: 128 rows x 8
      int r = idx >> 3, c4 = idx & 7;
      a4[i] = *(const float4*)&A[(size_t)(n0 + r) * K + k0 + c4 * 4];
    }
#pragma unroll
    for (int i = 0; i < 2; i++) {
      int idx = i * 256 + tid;           // 512 float4s: 32 k x 16
      int r = idx >> 4, c4 = idx & 15;
      w4[i] = *(const float4*)&W[(size_t)(k0 + r) * M + m0 + c4 * 4];
    }
  };
  // Convert + store staged registers into LDS buffer `buf`.
  auto lstore = [&](int buf) {
#pragma unroll
    for (int i = 0; i < 4; i++) {
      int idx = i * 256 + tid;
      int r = idx >> 3, c4 = idx & 7;
      *(v4h*)&As[buf][r * 40 + c4 * 4] = cvt4(a4[i]);
    }
#pragma unroll
    for (int i = 0; i < 2; i++) {
      int idx = i * 256 + tid;
      int r = idx >> 4, c4 = idx & 15;
      Bt[buf][(c4 * 4 + 0) * 40 + r] = (_Float16)w4[i].x;
      Bt[buf][(c4 * 4 + 1) * 40 + r] = (_Float16)w4[i].y;
      Bt[buf][(c4 * 4 + 2) * 40 + r] = (_Float16)w4[i].z;
      Bt[buf][(c4 * 4 + 3) * 40 + r] = (_Float16)w4[i].w;
    }
  };

  gload(0);
  lstore(0);
  __syncthreads();

  const int nk = K >> 5;
  for (int t = 0; t < nk; t++) {
    const int p = t & 1;
    if (t + 1 < nk) {
      gload((t + 1) << 5);               // issue next slab; latency hides
      if (t + 2 < nk) {                  // and prefetch the slab after that
        __builtin_prefetch(&A[(size_t)(n0 + (tid >> 1)) * K + ((t + 2) << 5)], 0, 0);
        __builtin_prefetch(&W[(size_t)(((t + 2) << 5) + wv) * M + m0 + lane], 0, 0);
      }
    }
    // Compute slab t from LDS buffer p: 2x ds_load_b128 per operand fragment.
    const _Float16* asb = As[p];
    const _Float16* btb = Bt[p];
    v8h alo = *(const v8h*)&asb[arow * 40 + akoff];
    v8h ahi = *(const v8h*)&asb[arow * 40 + 8 + akoff];
    v16h af = cat16(alo, ahi);
#pragma unroll
    for (int ct = 0; ct < 4; ct++) {
      const _Float16* bp = &btb[(ct * 16 + bcol) * 40 + bkoff];
      v16h bf = cat16(*(const v8h*)bp, *(const v8h*)(bp + 8));
      acc[ct] = __builtin_amdgcn_wmma_f32_16x16x32_f16(
          false, af, false, bf, (short)0, acc[ct], false, false);
    }
    if (t + 1 < nk) lstore(1 - p);       // drain staged registers into LDS
    __syncthreads();
  }

  // Epilogue: C-matrix layout (lanes 0-15 -> rows 0..7, lanes 16-31 -> 8..15)
  const int mrow0 = (lane < 16) ? 0 : 8;
#pragma unroll
  for (int ct = 0; ct < 4; ct++) {
    int gc = m0 + ct * 16 + bcol;
    float bv = bias ? bias[gc] : 0.0f;
#pragma unroll
    for (int i = 0; i < 8; i++) {
      int gr = n0 + wv * 16 + mrow0 + i;
      float out = acc[ct][i] + bv;
      if (mode == 1) out += res[(size_t)gr * M + gc];
      if (mode == 2) out = fmaxf(out, 0.0f);
      C[(size_t)gr * M + gc] = out;
    }
  }
}

// ---------------------------------------------------------------------------
// Attention: one wave (32 threads) per (b, h, 16-query tile).
//   scores = (q*scale + u_h) K^T  [WMMA, K=DH=32]
//          + RBF bias (heads 0..3 only, on-the-fly from bond_matrix)
//   softmax over keys (wave-shuffle reductions)
//   ctx    = attn @ V             [WMMA]
// ---------------------------------------------------------------------------
__global__ void attn_kernel(const float* __restrict__ qbuf,
                            const float* __restrict__ kbuf,
                            const float* __restrict__ vbuf,
                            const float* __restrict__ bond,
                            const int* __restrict__ lengths,
                            const float* __restrict__ uvec,
                            const float* __restrict__ vvec,
                            float* __restrict__ ctx) {
  const int q0 = blockIdx.x * 16;
  const int h  = blockIdx.y;
  const int b  = blockIdx.z;
  const int lane = threadIdx.x;

  __shared__ float qu[16 * 32];
  __shared__ float qv[16 * 32];
  __shared__ float sc[16 * 256];

  const int len = lengths[b];
  const float scale = 0.17677669529663687f;  // 1/sqrt(32)

  // Load and pre-bias the query tile.
#pragma unroll
  for (int i = 0; i < 16; i++) {
    int idx = i * 32 + lane;
    int r = idx >> 5, c = idx & 31;
    float qraw = qbuf[(size_t)(b * TT + q0 + r) * DD + h * DHH + c] * scale;
    qu[idx] = qraw + uvec[h * DHH + c];
    qv[idx] = qraw + vvec[h * DHH + c];
  }
  __syncthreads();

  const int arow = lane & 15;
  const int akoff = (lane < 16) ? 0 : 16;
  const int bcol = lane & 15;
  const int bkoff = (lane < 16) ? 0 : 16;

  // a_c = (q*scale + u) K^T, one WMMA per 16-key tile (K dim = DH = 32).
  v16h afrag;
#pragma unroll
  for (int e = 0; e < 8; e++) {
    afrag[e]     = (_Float16)qu[arow * 32 + akoff + e];
    afrag[e + 8] = (_Float16)qu[arow * 32 + 8 + akoff + e];
  }
  for (int kt = 0; kt < 16; kt++) {
    const int key0 = kt * 16;
    const float4* krow = (const float4*)
        &kbuf[(size_t)(b * TT + key0 + bcol) * DD + h * DHH + bkoff];
    float4 kq[4] = {krow[0], krow[1], krow[2], krow[3]};
    v16h bfrag;
#pragma unroll
    for (int j = 0; j < 4; j++) {
      bfrag[j * 4 + 0] = (_Float16)kq[j].x;
      bfrag[j * 4 + 1] = (_Float16)kq[j].y;
      bfrag[j * 4 + 2] = (_Float16)kq[j].z;
      bfrag[j * 4 + 3] = (_Float16)kq[j].w;
    }
    v8f acc;
    for (int i = 0; i < 8; i++) acc[i] = 0.0f;
    acc = __builtin_amdgcn_wmma_f32_16x16x32_f16(
        false, afrag, false, bfrag, (short)0, acc, false, false);
    const int m0 = (lane < 16) ? 0 : 8;
#pragma unroll
    for (int i = 0; i < 8; i++) sc[(m0 + i) * 256 + key0 + bcol] = acc[i];
  }
  __syncthreads();

  // Add RBF relative bias (heads 0..3) and key-padding mask.
  const float inv_gap = 20.0f;              // 1 / 0.05
  const float delta = 6.4f / 127.0f;        // CENTERS spacing
  for (int i = 0; i < 128; i++) {
    int idx = i * 32 + lane;                // covers 16x256 scores
    int r = idx >> 8, k = idx & 255;
    float s = sc[idx];
    if (k >= len) { sc[idx] = -1e18f; continue; }
    if (h < 4) {
      bool qpad = (q0 + r) >= len;
      float dm = qpad ? 1e9f : bond[((size_t)(b * TT + q0 + r)) * TT + k];
      if (dm < 100.0f) {                    // else every RBF term underflows to 0
        float bd = 0.0f;
#pragma unroll
        for (int d = 0; d < 32; d++) {
          float cj = (float)(h * DHH + d) * delta;
          float df = dm - cj;
          bd += qv[r * 32 + d] * __expf(-df * df * inv_gap);
        }
        s += bd;
      }
    }
    sc[idx] = s;
  }
  __syncthreads();

  // Softmax over keys, one row at a time, wave-shuffle reductions (wave32).
  for (int r = 0; r < 16; r++) {
    float m = -1e30f;
    for (int k = lane; k < 256; k += 32) m = fmaxf(m, sc[r * 256 + k]);
    for (int off = 16; off > 0; off >>= 1) m = fmaxf(m, __shfl_xor(m, off, 32));
    float sum = 0.0f;
    for (int k = lane; k < 256; k += 32) {
      float e = __expf(sc[r * 256 + k] - m);
      sc[r * 256 + k] = e;
      sum += e;
    }
    for (int off = 16; off > 0; off >>= 1) sum += __shfl_xor(sum, off, 32);
    float inv = 1.0f / sum;
    for (int k = lane; k < 256; k += 32) sc[r * 256 + k] *= inv;
  }
  __syncthreads();

  // ctx = attn[16x256] @ V[256x32]: 8 K-steps x 2 column tiles.
  v8f cacc0, cacc1;
  for (int i = 0; i < 8; i++) { cacc0[i] = 0.0f; cacc1[i] = 0.0f; }
  for (int k0 = 0; k0 < 256; k0 += 32) {
    v16h pa;
#pragma unroll
    for (int e = 0; e < 8; e++) {
      pa[e]     = (_Float16)sc[arow * 256 + k0 + akoff + e];
      pa[e + 8] = (_Float16)sc[arow * 256 + k0 + 8 + akoff + e];
    }
    v16h pb0, pb1;
#pragma unroll
    for (int e = 0; e < 16; e++) {
      const float* vr = &vbuf[(size_t)(b * TT + k0 + bkoff + e) * DD + h * DHH];
      pb0[e] = (_Float16)vr[bcol];
      pb1[e] = (_Float16)vr[16 + bcol];
    }
    cacc0 = __builtin_amdgcn_wmma_f32_16x16x32_f16(
        false, pa, false, pb0, (short)0, cacc0, false, false);
    cacc1 = __builtin_amdgcn_wmma_f32_16x16x32_f16(
        false, pa, false, pb1, (short)0, cacc1, false, false);
  }
  const int m0 = (lane < 16) ? 0 : 8;
#pragma unroll
  for (int i = 0; i < 8; i++) {
    size_t grow = (size_t)(b * TT + q0 + m0 + i) * DD + h * DHH;
    ctx[grow + bcol] = cacc0[i];
    ctx[grow + 16 + bcol] = cacc1[i];
  }
}

// ---------------------------------------------------------------------------
// Host-side orchestration.
// ---------------------------------------------------------------------------
extern "C" void kernel_launch(void* const* d_in, const int* in_sizes, int n_in,
                              void* d_out, int out_size, void* d_ws, size_t ws_size,
                              hipStream_t stream) {
  (void)in_sizes; (void)n_in; (void)out_size; (void)ws_size;

  const int*   src     = (const int*)d_in[0];
  const int*   lengths = (const int*)d_in[1];
  const float* bond    = (const float*)d_in[2];
  // d_in[3] timestep, d_in[5..8] timestep-MLP: provably dead (softmax shift
  // invariance over the key axis for heads 4..7) -> never read.
  const float* emb   = (const float*)d_in[4];
  const float* uvec  = (const float*)d_in[9];
  const float* vvec  = (const float*)d_in[10];
  const float* Wq = (const float*)d_in[11]; const float* bq = (const float*)d_in[12];
  const float* Wk = (const float*)d_in[13]; const float* bk = (const float*)d_in[14];
  const float* Wv = (const float*)d_in[15]; const float* bv = (const float*)d_in[16];
  const float* Wo = (const float*)d_in[17]; const float* bo = (const float*)d_in[18];
  const float* g1 = (const float*)d_in[19]; const float* be1 = (const float*)d_in[20];
  const float* g2 = (const float*)d_in[21]; const float* be2 = (const float*)d_in[22];
  const float* fw1 = (const float*)d_in[23]; const float* fb1 = (const float*)d_in[24];
  const float* fw2 = (const float*)d_in[25]; const float* fb2 = (const float*)d_in[26];
  const float* lnfg = (const float*)d_in[27]; const float* lnfb = (const float*)d_in[28];

  float* ws = (float*)d_ws;
  const size_t ND = (size_t)NN * DD;     // 524288
  float* xA = ws;                        // residual stream A
  float* xB = xA + ND;                   // residual stream B
  float* hb = xB + ND;                   // LN output
  float* qb = hb + ND;
  float* kb = qb + ND;
  float* vb = kb + ND;
  float* cb = vb + ND;                   // attention context
  float* ff = cb + ND;                   // FFN hidden, N x DFF (2M floats)

  embed_kernel<<<NN, DD, 0, stream>>>(src, emb, xA);

  const dim3 gProj(4, 16);               // M=256: 4 col tiles, N=2048: 16 row tiles
  const dim3 gFF1(16, 16);               // M=1024
  const dim3 gAttn(TT / 16, HH, BB);

  for (int l = 0; l < 4; l++) {
    const size_t wOff = (size_t)l * DD * DD;
    const size_t dOff = (size_t)l * DD;
    const size_t f1Off = (size_t)l * DD * DFFF;
    const size_t f2Off = (size_t)l * DFFF * DD;
    const size_t fbOff = (size_t)l * DFFF;

    ln_kernel<<<NN, DD, 0, stream>>>(xA, g1 + dOff, be1 + dOff, hb);
    gemm_kernel<<<gProj, 256, 0, stream>>>(hb, Wq + wOff, bq + dOff, nullptr, qb, NN, DD, DD, 0);
    gemm_kernel<<<gProj, 256, 0, stream>>>(hb, Wk + wOff, bk + dOff, nullptr, kb, NN, DD, DD, 0);
    gemm_kernel<<<gProj, 256, 0, stream>>>(hb, Wv + wOff, bv + dOff, nullptr, vb, NN, DD, DD, 0);
    attn_kernel<<<gAttn, 32, 0, stream>>>(qb, kb, vb, bond, lengths, uvec, vvec, cb);
    gemm_kernel<<<gProj, 256, 0, stream>>>(cb, Wo + wOff, bo + dOff, xA, xB, NN, DD, DD, 1);
    ln_kernel<<<NN, DD, 0, stream>>>(xB, g2 + dOff, be2 + dOff, hb);
    gemm_kernel<<<gFF1, 256, 0, stream>>>(hb, fw1 + f1Off, fb1 + fbOff, nullptr, ff, NN, DD, DFFF, 2);
    gemm_kernel<<<gProj, 256, 0, stream>>>(ff, fw2 + f2Off, fb2 + dOff, xB, xA, NN, DFFF, DD, 1);
  }

  ln_kernel<<<NN, DD, 0, stream>>>(xA, lnfg, lnfb, (float*)d_out);
}
